// Layer_87041807221059
// MI455X (gfx1250) — matrix operand order repaired and checked by reading, not compiled, not run on hardware
//
#include <hip/hip_runtime.h>
#include <hip/hip_bf16.h>

// out[b][o] = sum_k X[b][k] * Wcat[o][k],  X = [x0|x1] (B x 256, f16-converted),
// Wcat = [w0*m0 | w1*m1] (128 x 256, f16). fp32 accumulation via WMMA.

typedef _Float16 v16h __attribute__((ext_vector_type(16)));
typedef _Float16 v8h  __attribute__((ext_vector_type(8)));
typedef _Float16 v4h  __attribute__((ext_vector_type(4)));
typedef float    v8f  __attribute__((ext_vector_type(8)));
typedef float    v4f  __attribute__((ext_vector_type(4)));

#define DIN    128          // per-input feature dim
#define DOUT   128
#define KTOT   256          // concatenated K
#define M_BLK  128          // rows per block (8 waves x 16)
#define BLK_T  256          // threads per block (8 wave32s)
#define NT     8            // 16-col N tiles per wave (covers all 128 outputs)
#define XS     264          // padded LDS row stride in halves (256 + 8)

// ---------------- prep: Wcat[o][k] = (k<128 ? w0*m0 : w1*m1) as f16 ----------
__global__ __launch_bounds__(256) void prep_kernel(
    const float* __restrict__ w0, const float* __restrict__ w1,
    const float* __restrict__ m0, const float* __restrict__ m1,
    _Float16* __restrict__ wcat)
{
    int idx = blockIdx.x * 256 + threadIdx.x;   // 0 .. 32767
    int o = idx >> 8;
    int k = idx & 255;
    float v;
    if (k < DIN) v = w0[o * DIN + k] * m0[o * DIN + k];
    else         v = w1[o * DIN + (k - DIN)] * m1[o * DIN + (k - DIN)];
    wcat[o * KTOT + k] = (_Float16)v;
}

// ---------------- main GEMM ------------------------------------------------
extern __shared__ _Float16 smem[];

__global__ __launch_bounds__(BLK_T) void gemm_kernel(
    const float* __restrict__ x0, const float* __restrict__ x1,
    const _Float16* __restrict__ wcat, float* __restrict__ out)
{
    _Float16* xs = smem;                 // [M_BLK][XS]  f16 X slab
    _Float16* ws = smem + M_BLK * XS;    // [DOUT][XS]   f16 Wcat

    const int tid  = threadIdx.x;
    const int lane = tid & 31;
    const int wave = tid >> 5;
    const long rowBase = (long)blockIdx.x * M_BLK;

    // ---- stage X: 128 rows x 256 fp32 cols -> f16 LDS (coalesced b128 loads)
    #pragma unroll
    for (int i = 0; i < 32; ++i) {
        int idx = i * BLK_T + tid;           // 0 .. 8191 float4 tiles
        int r   = idx >> 6;                  // 64 float4 per row
        int c4  = (idx & 63) << 2;           // col base (multiple of 4)
        const float* src = (c4 < DIN) ? (x0 + (rowBase + r) * DIN + c4)
                                      : (x1 + (rowBase + r) * DIN + (c4 - DIN));
        v4f v = *(const v4f*)src;
        v4h h;
        h.x = (_Float16)v.x; h.y = (_Float16)v.y;
        h.z = (_Float16)v.z; h.w = (_Float16)v.w;
        *(v4h*)(xs + r * XS + c4) = h;       // 8B-aligned ds_store_b64
    }
    // ---- stage Wcat: 128 x 256 f16 (16B chunks)
    #pragma unroll
    for (int i = 0; i < 16; ++i) {
        int idx = i * BLK_T + tid;           // 0 .. 4095 v8h tiles
        int r   = idx >> 5;                  // 32 v8h per row
        int c8  = (idx & 31) << 3;
        *(v8h*)(ws + r * XS + c8) = *(const v8h*)(wcat + r * KTOT + c8);
    }
    __syncthreads();

    const int half = lane >> 4;
    const int l15  = lane & 15;
    const int mrow = wave * 16 + l15;        // local A row for this lane

    v8f acc[NT] = {};

    #pragma unroll
    for (int kc = 0; kc < KTOT / 32; ++kc) {
        // A fragment, 16-bit 16x32 layout: lanes 0-15 M=0..15;
        // a[0..7] = K 8*half+{0..7}, a[8..15] = K 16+8*half+{0..7}
        const _Float16* xrow = xs + mrow * XS + kc * 32 + 8 * half;
        v8h alo = *(const v8h*)(xrow);        // ds_load_b128
        v8h ahi = *(const v8h*)(xrow + 16);   // ds_load_b128
        v16h a = __builtin_shufflevector(alo, ahi,
                 0,1,2,3,4,5,6,7,8,9,10,11,12,13,14,15);

        const int kb = kc * 32 + 8 * half;    // B: b[0..7] = K 8*half+{0..7}, b[8..15] = K 16+8*half+{0..7}
        #pragma unroll
        for (int n = 0; n < NT; ++n) {
            const _Float16* wrow = ws + (n * 16 + l15) * XS + kb;
            v8h blo = *(const v8h*)(wrow);
            v8h bhi = *(const v8h*)(wrow + 16);
            v16h b = __builtin_shufflevector(blo, bhi,
                     0,1,2,3,4,5,6,7,8,9,10,11,12,13,14,15);
            acc[n] = __builtin_amdgcn_wmma_f32_16x16x32_f16(
                         /*neg_a=*/false, a, /*neg_b=*/false, b,
                         /*c_mod=*/(short)0, acc[n],
                         /*reuse_a=*/false, /*reuse_b=*/false);
        }
    }

    // ---- store D: VGPR r -> M = 8*half + r, N = l15 (per 16x16 f32 layout)
    #pragma unroll
    for (int n = 0; n < NT; ++n) {
        float* obase = out + (rowBase + wave * 16 + half * 8) * DOUT + n * 16 + l15;
        #pragma unroll
        for (int r = 0; r < 8; ++r)
            obase[(long)r * DOUT] = acc[n][r];
    }
}

extern "C" void kernel_launch(void* const* d_in, const int* in_sizes, int n_in,
                              void* d_out, int out_size, void* d_ws, size_t ws_size,
                              hipStream_t stream) {
    const float* x0 = (const float*)d_in[0];
    const float* x1 = (const float*)d_in[1];
    const float* w0 = (const float*)d_in[2];
    const float* w1 = (const float*)d_in[3];
    const float* m0 = (const float*)d_in[4];
    const float* m1 = (const float*)d_in[5];
    _Float16* wcat  = (_Float16*)d_ws;          // 128*256 f16 = 64 KB scratch
    float* out      = (float*)d_out;

    const int Btot = in_sizes[0] / DIN;         // 524288
    const int nblk = Btot / M_BLK;              // 4096

    // 1) fuse masks into f16 weights once per launch (deterministic)
    prep_kernel<<<(DOUT * KTOT) / 256, 256, 0, stream>>>(w0, w1, m0, m1, wcat);

    // 2) WMMA GEMM; 132 KB dynamic LDS (CDNA5 WGP has 320 KB)
    const size_t lds_bytes = (size_t)(M_BLK + DOUT) * XS * sizeof(_Float16); // 135168
    hipFuncSetAttribute((const void*)gemm_kernel,
                        hipFuncAttributeMaxDynamicSharedMemorySize,
                        (int)lds_bytes);
    gemm_kernel<<<nblk, BLK_T, lds_bytes, stream>>>(x0, x1, wcat, out);
}
